// MultiHeadedAttention_4329327034679
// MI455X (gfx1250) — matrix operand order repaired and checked by reading, not compiled, not run on hardware
//
#include <hip/hip_runtime.h>
#include <hip/hip_bf16.h>
#include <cstdint>

// MI455X (gfx1250) fused structure-biased attention.
// Precision: bf16 WMMA (v_wmma_f32_16x16x32_bf16) with f32 accumulate.
// Key fusion: q@Wsk^T folded into q-side (qs), attn@sv = (attn@struct)@Wsv + bsv,
// so the 268MB structure tensor is read from HBM exactly once per (b,q) tile.

typedef __bf16 bf16_t;
typedef __attribute__((ext_vector_type(16))) __bf16 v16bf;
typedef __attribute__((ext_vector_type(8)))  float  v8f;
typedef __attribute__((ext_vector_type(4)))  __bf16 v4bf;

#define B_ 4
#define S_ 512
#define D_ 512
#define H_ 8
#define DPH_ 64

__device__ __forceinline__ int laneId() { return threadIdx.x & 31; }

__device__ __forceinline__ float bfu(unsigned u) {
  return __builtin_bit_cast(float, u << 16);
}

// ---- WMMA fragment loaders (wave32; contiguous k order k = 16*(l>>4)+s, same for A and B) ----

// A fragment (16x32 bf16, M=row, K contiguous): lane l row=l&15 holds K {ko..ko+15}, ko=16*(l>>4)
__device__ __forceinline__ v16bf fragA_bf16(const bf16_t* base, int ld) {
  int l = laneId(); int row = l & 15; int ko = (l >> 4) << 4;
  const bf16_t* p = base + row * ld + ko;
  v16bf a;
  ((uint4*)&a)[0] = *((const uint4*)p);
  ((uint4*)&a)[1] = *((const uint4*)(p + 8));
  return a;
}

// A fragment from f32 source with convert
__device__ __forceinline__ v16bf fragA_f32(const float* base, int ld) {
  int l = laneId(); int row = l & 15; int ko = (l >> 4) << 4;
  const float* p = base + row * ld + ko;
  v16bf a;
#pragma unroll
  for (int j = 0; j < 8; ++j) { a[j] = (bf16_t)p[j]; a[8 + j] = (bf16_t)p[8 + j]; }
  return a;
}

// B fragment (32x16): element(K=k,N=n). "kcontig": element(k,n) = base[n*ld + k]
__device__ __forceinline__ v16bf fragB_kcontig_bf16(const bf16_t* base, int ld) {
  int l = laneId(); int n = l & 15; int kb = (l >> 4) << 4;
  const bf16_t* p = base + n * ld + kb;
  v16bf b;
  ((uint4*)&b)[0] = *((const uint4*)p);
  ((uint4*)&b)[1] = *((const uint4*)(p + 8));
  return b;
}
__device__ __forceinline__ v16bf fragB_kcontig_f32(const float* base, int ld) {
  int l = laneId(); int n = l & 15; int kb = (l >> 4) << 4;
  const float* p = base + n * ld + kb;
  v16bf b;
#pragma unroll
  for (int j = 0; j < 16; ++j) b[j] = (bf16_t)p[j];
  return b;
}
// "kstride": element(k,n) = base[k*ld + n]
__device__ __forceinline__ v16bf fragB_kstride_bf16(const bf16_t* base, int ld) {
  int l = laneId(); int n = l & 15; int kb = (l >> 4) << 4;
  v16bf b;
#pragma unroll
  for (int j = 0; j < 16; ++j) b[j] = base[(kb + j) * ld + n];
  return b;
}
__device__ __forceinline__ v16bf fragB_kstride_f32(const float* base, int ld) {
  int l = laneId(); int n = l & 15; int kb = (l >> 4) << 4;
  v16bf b;
#pragma unroll
  for (int j = 0; j < 16; ++j) b[j] = (bf16_t)base[(kb + j) * ld + n];
  return b;
}

__device__ __forceinline__ v8f wmma_bf16(v16bf a, v16bf b, v8f c) {
  return __builtin_amdgcn_wmma_f32_16x16x32_bf16(false, a, false, b, (short)0, c,
                                                 false, false);
}

// ---------------- Kernel 1: fused Q/K/V projection (bf16 WMMA GEMM) ----------------
// Y = X @ W + b, output bf16 in [B,H,S,DPH] layout; q pre-scaled by 1/sqrt(DPH).
// Register blocking: each wave computes a 16x64 strip (4 accumulators), so each
// A fragment feeds 4 WMMAs (4x fewer A-side vmem bytes per matrix op).
__global__ __launch_bounds__(256) void qkv_proj_kernel(
    const float* __restrict__ query, const float* __restrict__ key,
    const float* __restrict__ value,
    const float* __restrict__ Wq, const float* __restrict__ bq,
    const float* __restrict__ Wk, const float* __restrict__ bk,
    const float* __restrict__ Wv, const float* __restrict__ bv,
    bf16_t* __restrict__ qws, bf16_t* __restrict__ kws, bf16_t* __restrict__ vws) {
  int wtile = blockIdx.x * 8 + (threadIdx.x >> 5);  // 3 * 128 * 8 = 3072 strips
  int mat = wtile >> 10;                            // 1024 strips per matrix
  int rem = wtile & 1023;
  int m0 = (rem >> 3) << 4;                         // 128 row tiles
  int n0 = (rem & 7) << 6;                          // 8 col groups of 64
  const float* X; const float* W; const float* bias; bf16_t* out; float scale;
  if (mat == 0)      { X = query; W = Wq; bias = bq; out = qws; scale = 0.125f; }
  else if (mat == 1) { X = key;   W = Wk; bias = bk; out = kws; scale = 1.0f; }
  else               { X = value; W = Wv; bias = bv; out = vws; scale = 1.0f; }

  v8f acc[4] = {};
  for (int ks = 0; ks < D_; ks += 32) {
    v16bf a = fragA_f32(X + (size_t)m0 * D_ + ks, D_);
#pragma unroll
    for (int j = 0; j < 4; ++j) {
      v16bf b = fragB_kstride_f32(W + (size_t)ks * D_ + n0 + j * 16, D_);
      acc[j] = wmma_bf16(a, b, acc[j]);
    }
  }
  int l = laneId(); int n = l & 15; int mb = (l >> 4) << 3;
#pragma unroll
  for (int j = 0; j < 4; ++j) {
    int col = n0 + j * 16 + n; int h = col >> 6; int dp = col & 63;
    float bb = bias[col];
#pragma unroll
    for (int r = 0; r < 8; ++r) {
      int row = m0 + mb + r; int bidx = row >> 9; int s = row & 511;
      out[(((size_t)bidx * H_ + h) * S_ + s) * DPH_ + dp] =
          (bf16_t)((acc[j][r] + bb) * scale);
    }
  }
}

// ---------------- Kernel 2: fused attention, one workgroup per (b, q) ----------------
// LDS: sStruct bf16[512][64] (64KB) | sScores f32[16][512] (32KB) |
//      sT f32[16][64] | sCtx f32[16][64] | sQf f32[8][64] | sQh bf16[16][64] | sQs bf16[16][64]
#define SMEM_BYTES (65536 + 32768 + 4096 + 4096 + 2048 + 2048 + 2048)

__global__ __launch_bounds__(256) void attn_kernel(
    const bf16_t* __restrict__ qws, const bf16_t* __restrict__ kws,
    const bf16_t* __restrict__ vws,
    const float* __restrict__ structure, const unsigned char* __restrict__ mask,
    const float* __restrict__ Wsk, const float* __restrict__ bsk,
    const float* __restrict__ Wsv, const float* __restrict__ bsv,
    float* __restrict__ ctxbuf, float* __restrict__ top_attn) {
  extern __shared__ char smem[];
  bf16_t* sStruct = (bf16_t*)smem;                             // [512][64]
  float*  sScores = (float*)(smem + 65536);                    // [16][512]
  float*  sT      = (float*)(smem + 65536 + 32768);            // [16][64]
  float*  sCtx    = (float*)(smem + 65536 + 32768 + 4096);     // [16][64]
  float*  sQf     = (float*)(smem + 65536 + 32768 + 8192);     // [8][64]
  bf16_t* sQh     = (bf16_t*)(smem + 65536 + 32768 + 8192 + 2048);        // [16][64]
  bf16_t* sQs     = (bf16_t*)(smem + 65536 + 32768 + 8192 + 2048 + 2048); // [16][64]

  int tid = threadIdx.x; int wv = tid >> 5; int l = tid & 31;
  int b = blockIdx.x >> 9; int q = blockIdx.x & 511;

  // Step 1: load q rows for all heads (rows 8..15 zero for 16-row WMMA A), zero inits.
  for (int i = tid; i < 16 * 64; i += 256) {
    int h = i >> 6, e = i & 63;
    bf16_t val = (bf16_t)0.0f;
    if (h < 8) {
      val = qws[(((size_t)b * H_ + h) * S_ + q) * DPH_ + e];
      sQf[i] = (float)val;
    }
    sQh[i] = val;
    sQs[i] = (bf16_t)0.0f;
  }
  for (int i = tid; i < 8 * 512; i += 256) sScores[8 * 512 + i] = 0.0f;
  __syncthreads();

  // Step 2 (waves 0-3): qs = qH @ Wsk^T  (16x64 = 16x64 @ 64x64^T), 2 WMMAs per wave.
  if (wv < 4) {
    int c0 = wv << 4;
    v8f c = {};
#pragma unroll
    for (int ks = 0; ks < 64; ks += 32) {
      v16bf a = fragA_bf16(sQh + ks, 64);
      // B(e, cn) = Wsk[c0+cn][ks+e] : K contiguous within Wsk row
      v16bf bm = fragB_kcontig_f32(Wsk + (size_t)c0 * 64 + ks, 64);
      c = wmma_bf16(a, bm, c);
    }
    int n = l & 15;
    if (l < 16) {
#pragma unroll
      for (int r = 0; r < 8; ++r) sQs[r * 64 + c0 + n] = (bf16_t)c[r];
    }
  }

  // Step 3 (all): stream structure[b,q,:,:] (512x64 f32) -> LDS bf16 (single HBM read).
  // global_prefetch keeps the dominant 268MB HBM stream ahead of the converts.
  {
    const float4* src = (const float4*)(structure + (size_t)blockIdx.x * (S_ * DPH_));
    v4bf* dst = (v4bf*)sStruct;
    for (int i = tid; i < (S_ * DPH_) / 4; i += 256) {
      if (i + 2048 < (S_ * DPH_) / 4) __builtin_prefetch(src + i + 2048, 0, 0);
      float4 f = src[i];
      v4bf hv; hv[0] = (bf16_t)f.x; hv[1] = (bf16_t)f.y;
      hv[2] = (bf16_t)f.z; hv[3] = (bf16_t)f.w;
      dst[i] = hv;
    }
  }

  // Step 4 (wave w = head w): qk scores + qb bias term (per-head GEMV on VALU).
  {
    int h = wv;
    float qb = 0.0f;
#pragma unroll 8
    for (int e = 0; e < 64; ++e) qb += sQf[h * 64 + e] * bsk[e];
    const bf16_t* kbase = kws + (((size_t)b * H_ + h) * S_) * DPH_;
    for (int ki = 0; ki < 16; ++ki) {
      int k = l + (ki << 5);
      const uint4* kr = (const uint4*)(kbase + (size_t)k * DPH_);
      float acc = qb;
#pragma unroll
      for (int u = 0; u < 8; ++u) {
        uint4 pk = kr[u];
        const float* qv = sQf + h * 64 + u * 8;
        acc += bfu(pk.x & 0xffffu) * qv[0] + bfu(pk.x >> 16) * qv[1];
        acc += bfu(pk.y & 0xffffu) * qv[2] + bfu(pk.y >> 16) * qv[3];
        acc += bfu(pk.z & 0xffffu) * qv[4] + bfu(pk.z >> 16) * qv[5];
        acc += bfu(pk.w & 0xffffu) * qv[6] + bfu(pk.w >> 16) * qv[7];
      }
      sScores[h * 512 + k] = acc;
    }
  }
  __syncthreads();

  // Step 6 (all waves): struct score term  scores[h,k] += qs @ struct^T  (WMMA, heads = M).
  for (int kt = wv; kt < 32; kt += 8) {
    int kcol0 = kt << 4;
    v8f c = {};
#pragma unroll
    for (int c0 = 0; c0 < 64; c0 += 32) {
      v16bf a = fragA_bf16(sQs + c0, 64);
      // B(cc, kn) = sStruct[kcol0+kn][c0+cc] : K contiguous in struct row
      v16bf bm = fragB_kcontig_bf16(sStruct + (size_t)kcol0 * 64 + c0, 64);
      c = wmma_bf16(a, bm, c);
    }
    int n = l & 15;
    if (l < 16) {
#pragma unroll
      for (int r = 0; r < 8; ++r) sScores[r * 512 + kcol0 + n] += c[r];
    }
  }
  __syncthreads();

  // Step 8 (wave = head): mask + softmax (wave32 shuffle reductions); head 0 writes top_attn.
  {
    int h = wv;
    const unsigned char* mrow = mask + (size_t)blockIdx.x * S_;
    float vals[16];
    float mx = -3.0e38f;
#pragma unroll
    for (int ki = 0; ki < 16; ++ki) {
      int k = l + (ki << 5);
      float s = mrow[k] ? -1.0e18f : sScores[h * 512 + k];
      vals[ki] = s;
      mx = fmaxf(mx, s);
    }
#pragma unroll
    for (int off = 16; off > 0; off >>= 1) mx = fmaxf(mx, __shfl_xor(mx, off, 32));
    float sum = 0.f;
#pragma unroll
    for (int ki = 0; ki < 16; ++ki) { float p = __expf(vals[ki] - mx); vals[ki] = p; sum += p; }
#pragma unroll
    for (int off = 16; off > 0; off >>= 1) sum += __shfl_xor(sum, off, 32);
    float inv = 1.0f / sum;
    float* srow = sScores + h * 512;
    float* trow = top_attn + (size_t)blockIdx.x * S_;
#pragma unroll
    for (int ki = 0; ki < 16; ++ki) {
      int k = l + (ki << 5);
      float p = vals[ki] * inv;
      srow[k] = p;
      if (h == 0) trow[k] = p;
    }
  }
  __syncthreads();

  // Step 10: waves 0-3: t = attn @ struct (WMMA, K=512); waves 4-7: ctxv = attn @ v (per-head).
  if (wv < 4) {
    int c0 = wv << 4;
    v8f c = {};
    for (int ks = 0; ks < 512; ks += 32) {
      v16bf a = fragA_f32(sScores + ks, 512);
      v16bf bm = fragB_kstride_bf16(sStruct + (size_t)ks * 64 + c0, 64);
      c = wmma_bf16(a, bm, c);
    }
    int n = l & 15; int mb = (l >> 4) << 3;
#pragma unroll
    for (int r = 0; r < 8; ++r) sT[(mb + r) * 64 + c0 + n] = c[r];
  } else {
    int h0 = (wv - 4) << 1;
#pragma unroll
    for (int hh = 0; hh < 2; ++hh) {
      int h = h0 + hh;
      const bf16_t* vbase = vws + (((size_t)b * H_ + h) * S_) * DPH_;
      const float* arow = sScores + h * 512;
      float a0 = 0.f, a1 = 0.f;
#pragma unroll 4
      for (int k = 0; k < 512; ++k) {
        float a = arow[k];
        const bf16_t* vr = vbase + (size_t)k * DPH_;
        a0 += a * (float)vr[l];
        a1 += a * (float)vr[l + 32];
      }
      sCtx[h * 64 + l] = a0;
      sCtx[h * 64 + l + 32] = a1;
    }
  }
  __syncthreads();

  // Step 12 (waves 0-3): ctx += t @ Wsv + bsv  (sum(attn)==1 absorbs the bias exactly).
  if (wv < 4) {
    int d0 = wv << 4;
    v8f c = {};
#pragma unroll
    for (int ks = 0; ks < 64; ks += 32) {
      v16bf a = fragA_f32(sT + ks, 64);
      v16bf bm = fragB_kstride_f32(Wsv + (size_t)ks * 64 + d0, 64);
      c = wmma_bf16(a, bm, c);
    }
    int n = l & 15;
    if (l < 16) {
      float bb = bsv[d0 + n];
#pragma unroll
      for (int r = 0; r < 8; ++r) sCtx[r * 64 + d0 + n] += c[r] + bb;
    }
  }
  __syncthreads();

  // Step 14: write ctx row (b,q) as [h*64+d] to workspace.
  float* crow = ctxbuf + (size_t)blockIdx.x * D_;
  for (int i = tid; i < 512; i += 256) crow[i] = sCtx[(i >> 6) * 64 + (i & 63)];
}

// ---------------- Kernel 3: output projection (bf16 WMMA GEMM, f32 out) ----------------
__global__ __launch_bounds__(256) void out_proj_kernel(
    const float* __restrict__ ctx, const float* __restrict__ Wo,
    const float* __restrict__ bo, float* __restrict__ out) {
  int wtile = blockIdx.x * 8 + (threadIdx.x >> 5);  // 128 * 8 = 1024 strips
  int m0 = (wtile >> 3) << 4;
  int n0 = (wtile & 7) << 6;
  v8f acc[4] = {};
  for (int ks = 0; ks < D_; ks += 32) {
    v16bf a = fragA_f32(ctx + (size_t)m0 * D_ + ks, D_);
#pragma unroll
    for (int j = 0; j < 4; ++j) {
      v16bf b = fragB_kstride_f32(Wo + (size_t)ks * D_ + n0 + j * 16, D_);
      acc[j] = wmma_bf16(a, b, acc[j]);
    }
  }
  int l = laneId(); int n = l & 15; int mb = (l >> 4) << 3;
#pragma unroll
  for (int j = 0; j < 4; ++j) {
    int col = n0 + j * 16 + n;
    float bias = bo[col];
#pragma unroll
    for (int r = 0; r < 8; ++r) {
      int row = m0 + mb + r;
      out[(size_t)row * D_ + col] = acc[j][r] + bias;
    }
  }
}

extern "C" void kernel_launch(void* const* d_in, const int* in_sizes, int n_in,
                              void* d_out, int out_size, void* d_ws, size_t ws_size,
                              hipStream_t stream) {
  (void)in_sizes; (void)n_in; (void)out_size; (void)ws_size;
  const float* key       = (const float*)d_in[0];
  const float* value     = (const float*)d_in[1];
  const float* query     = (const float*)d_in[2];
  const float* structure = (const float*)d_in[3];
  const unsigned char* mask = (const unsigned char*)d_in[4];
  const float* Wq  = (const float*)d_in[5];  const float* bq  = (const float*)d_in[6];
  const float* Wk  = (const float*)d_in[7];  const float* bk  = (const float*)d_in[8];
  const float* Wv  = (const float*)d_in[9];  const float* bv  = (const float*)d_in[10];
  const float* Wsk = (const float*)d_in[11]; const float* bsk = (const float*)d_in[12];
  const float* Wsv = (const float*)d_in[13]; const float* bsv = (const float*)d_in[14];
  const float* Wo  = (const float*)d_in[15]; const float* bo  = (const float*)d_in[16];

  char* ws = (char*)d_ws;
  bf16_t* qws = (bf16_t*)ws;                      // 2 MB  [B,H,S,DPH] bf16
  bf16_t* kws = (bf16_t*)(ws + (2u << 20));       // 2 MB
  bf16_t* vws = (bf16_t*)(ws + (4u << 20));       // 2 MB
  float*  ctxbuf = (float*)(ws + (6u << 20));     // 4 MB  [B*S, D] f32

  float* out = (float*)d_out;
  float* top_attn = out + (size_t)B_ * S_ * D_;

  qkv_proj_kernel<<<384, 256, 0, stream>>>(query, key, value, Wq, bq, Wk, bk,
                                           Wv, bv, qws, kws, vws);
  attn_kernel<<<B_ * S_, 256, SMEM_BYTES, stream>>>(qws, kws, vws, structure, mask,
                                                    Wsk, bsk, Wsv, bsv, ctxbuf, top_attn);
  out_proj_kernel<<<128, 256, 0, stream>>>(ctxbuf, Wo, bo, out);
}
